// ContrastiveLoss_86629490360547
// MI455X (gfx1250) — matrix operand order repaired and checked by reading, not compiled, not run on hardware
//
#include <hip/hip_runtime.h>

// ---------------------------------------------------------------------------
// ContrastiveLoss on MI455X (gfx1250), wave32 + WMMA bf16x3 split-precision,
// with CDNA5 async global->LDS tile staging (ASYNCcnt path).
//
//   G[i,:] = feature[i,:] / (||feature[i]|| * sqrt(TAL))   (so ex = G G^T)
//   ex max per column is the diagonal = 1/TAL  -> constant-max logsumexp
//   loss = (1/n) * sum_j [ C + log(sum_i exp(ex[i,j]-C)) - P_j / npos_j ]
// ---------------------------------------------------------------------------

typedef __bf16 bf16_t;
typedef __attribute__((ext_vector_type(16))) __bf16 v16bf;
typedef __attribute__((ext_vector_type(8)))  __bf16 v8bf;
typedef __attribute__((ext_vector_type(8)))  float  v8f;

#define N_ROWS  8192
#define DIMF    1024
#define TAL     0.07f
#define CMAX    (1.0f / 0.07f)
#define BM      64            // M-tile staged in LDS per block
#define BK      128           // K-chunk staged in LDS
#define LDK     (BK + 8)      // pad: row stride 272B -> no DS bank conflicts
#define MSPLIT  16            // M-dimension split across blockIdx.y
#define ROWS_PER_SLICE (N_ROWS / MSPLIT)   // 512

__device__ __forceinline__ v8f wmma_bf16(v16bf a, v16bf b, v8f c) {
  // v_wmma_f32_16x16x32_bf16  D = A x B + C
  return __builtin_amdgcn_wmma_f32_16x16x32_bf16(
      /*neg_a=*/false, a, /*neg_b=*/false, b,
      /*c_mod=*/(short)0, c, /*reuse_a=*/false, /*reuse_b=*/false);
}

// A fragment (16-bit 16x32): per lane, elements 0..7 = K[off..off+7],
// elements 8..15 = K[off+16..off+23]  (off = 8 for lanes 16..31).
__device__ __forceinline__ v16bf load_a_frag(const bf16_t* p) {
  v8bf lo8 = *(const v8bf*)p;
  v8bf hi8 = *(const v8bf*)(p + 16);
  v16bf r;
#pragma unroll
  for (int i = 0; i < 8; ++i) { r[i] = lo8[i]; r[i + 8] = hi8[i]; }
  return r;
}

// CDNA5 async copy: 16B global -> LDS, no VGPR staging, tracked by ASYNCcnt.
// lds = workgroup-relative LDS byte offset (low 32 bits of generic pointer).
__device__ __forceinline__ void async_copy_b128(unsigned lds, const void* g) {
  asm volatile("global_load_async_to_lds_b128 %0, %1, off"
               :: "v"(lds), "v"((unsigned long long)g)
               : "memory");
}
__device__ __forceinline__ void wait_asynccnt0() {
  asm volatile("s_wait_asynccnt 0x0" ::: "memory");
}
__device__ __forceinline__ unsigned lds_off(const void* p) {
  return (unsigned)(unsigned long long)p;   // LDS aperture: offset in [31:0]
}

// ---------------------------------------------------------------------------
// Kernel 1: row L2 norm, scale, split fp32 -> (bf16 hi, bf16 lo residual)
// ---------------------------------------------------------------------------
__global__ __launch_bounds__(256) void k_scale_split(
    const float* __restrict__ F, bf16_t* __restrict__ Ghi,
    bf16_t* __restrict__ Glo) {
  const int row = blockIdx.x;
  const float* fr = F + (size_t)row * DIMF;
  const int t = threadIdx.x;
  float x[4];
#pragma unroll
  for (int c = 0; c < 4; ++c) x[c] = fr[t + c * 256];
  float ss = x[0]*x[0] + x[1]*x[1] + x[2]*x[2] + x[3]*x[3];
#pragma unroll
  for (int o = 16; o > 0; o >>= 1) ss += __shfl_down(ss, o, 32);
  __shared__ float red[8];
  const int lane = t & 31, w = t >> 5;
  if (lane == 0) red[w] = ss;
  __syncthreads();
  float tot = 0.f;
#pragma unroll
  for (int i = 0; i < 8; ++i) tot += red[i];
  const float scale = 1.0f / sqrtf(tot * TAL);
#pragma unroll
  for (int c = 0; c < 4; ++c) {
    float v = x[c] * scale;
    bf16_t h = (bf16_t)v;
    bf16_t l = (bf16_t)(v - (float)h);
    Ghi[(size_t)row * DIMF + t + c * 256] = h;
    Glo[(size_t)row * DIMF + t + c * 256] = l;
  }
}

// ---------------------------------------------------------------------------
// Kernel 2: tiled Gram + fused streaming epilogue.
// grid = (64 column-tiles of 128, MSPLIT m-slices); block = 256 (8 waves).
// Each wave owns 16 columns exclusively -> deterministic, no atomics.
// A tiles staged in LDS with global_load_async_to_lds_b128.
// ---------------------------------------------------------------------------
__global__ __launch_bounds__(256, 1) void k_gram_epilogue(
    const bf16_t* __restrict__ Ghi, const bf16_t* __restrict__ Glo,
    const int* __restrict__ lab, float* __restrict__ Spart,
    float* __restrict__ Ppart, float* __restrict__ Cpart) {
  __shared__ bf16_t Ahi[BM][LDK];
  __shared__ bf16_t Alo[BM][LDK];

  const int tid   = threadIdx.x;
  const int lane  = tid & 31;
  const int wave  = tid >> 5;
  const int hsel  = lane >> 4;          // half-wave select (0/1)
  const int l16   = lane & 15;
  const int jcol  = blockIdx.x * 128 + wave * 16 + l16;   // owned column
  const int ms    = blockIdx.y;
  const int akoff = hsel * 8;           // A fragment K offset
  const int bkoff = hsel * 8;           // B fragment K offset
  const int labj  = lab[jcol];
  const bf16_t* browh = Ghi + (size_t)jcol * DIMF;
  const bf16_t* browl = Glo + (size_t)jcol * DIMF;

  // cooperative A-tile mapping: 4 threads per row, 32 K (64B) each
  const int lrow = tid >> 2;
  const int lseg = (tid & 3) * 32;
  const unsigned dsth = lds_off(&Ahi[lrow][lseg]);
  const unsigned dstl = lds_off(&Alo[lrow][lseg]);

  float s_acc = 0.f, p_acc = 0.f, n_acc = 0.f;

  const int m0 = ms * ROWS_PER_SLICE;
  for (int mt = m0; mt < m0 + ROWS_PER_SLICE; mt += BM) {
    v8f z = {};
    v8f acc[4];
#pragma unroll
    for (int s = 0; s < 4; ++s) acc[s] = z;

    for (int kc = 0; kc < DIMF; kc += BK) {
      __syncthreads();   // previous tile fully consumed
      {
        const bf16_t* sh = Ghi + (size_t)(mt + lrow) * DIMF + kc + lseg;
        const bf16_t* sl = Glo + (size_t)(mt + lrow) * DIMF + kc + lseg;
#pragma unroll
        for (int c = 0; c < 4; ++c) {
          async_copy_b128(dsth + c * 16, sh + c * 8);
          async_copy_b128(dstl + c * 16, sl + c * 8);
        }
      }
      wait_asynccnt0();  // own async copies landed in LDS
      __syncthreads();   // whole tile visible to all waves

#pragma unroll
      for (int ks = 0; ks < BK; ks += 32) {
        // B fragment: lane = column N, same K order as A fragment
        v16bf bh = load_a_frag(browh + kc + ks + bkoff);
        v16bf bl = load_a_frag(browl + kc + ks + bkoff);
        // preload all A fragments so ds_loads overlap the WMMA stream
        v16bf ah[4], al[4];
#pragma unroll
        for (int s = 0; s < 4; ++s) {
          ah[s] = load_a_frag(&Ahi[s * 16 + l16][ks + akoff]);
          al[s] = load_a_frag(&Alo[s * 16 + l16][ks + akoff]);
        }
#pragma unroll
        for (int s = 0; s < 4; ++s) {
          acc[s] = wmma_bf16(ah[s], bh, acc[s]);   // hi*hi
          acc[s] = wmma_bf16(ah[s], bl, acc[s]);   // hi*lo
          acc[s] = wmma_bf16(al[s], bh, acc[s]);   // lo*hi
        }
      }
    }

    // fused epilogue: C/D layout: lane -> N, VGPR r -> row r (+8 for hsel=1)
#pragma unroll
    for (int s = 0; s < 4; ++s) {
      const int ibase = mt + s * 16 + hsel * 8;
      int4 la = *(const int4*)(lab + ibase);
      int4 lb = *(const int4*)(lab + ibase + 4);
      int li[8] = {la.x, la.y, la.z, la.w, lb.x, lb.y, lb.z, lb.w};
#pragma unroll
      for (int r = 0; r < 8; ++r) {
        float e = acc[s][r];
        s_acc += __expf(e - CMAX);          // constant-max logsumexp term
        if (li[r] == labj) { p_acc += e; n_acc += 1.0f; }
      }
    }
  }

  // merge the two half-wave copies of each column, lanes 0..15 write out
  s_acc += __shfl_xor(s_acc, 16, 32);
  p_acc += __shfl_xor(p_acc, 16, 32);
  n_acc += __shfl_xor(n_acc, 16, 32);
  if (hsel == 0) {
    const int idx = ms * N_ROWS + jcol;
    Spart[idx] = s_acc;
    Ppart[idx] = p_acc;
    Cpart[idx] = n_acc;
  }
}

// ---------------------------------------------------------------------------
// Kernel 3: combine M-slice partials, per-column loss, global reduce.
// ---------------------------------------------------------------------------
__global__ __launch_bounds__(256) void k_finish(
    const float* __restrict__ Spart, const float* __restrict__ Ppart,
    const float* __restrict__ Cpart, float* __restrict__ out) {
  float tot = 0.f;
  for (int j = threadIdx.x; j < N_ROWS; j += 256) {
    float s = 0.f, p = 0.f, c = 0.f;
#pragma unroll
    for (int m = 0; m < MSPLIT; ++m) {
      s += Spart[m * N_ROWS + j];
      p += Ppart[m * N_ROWS + j];
      c += Cpart[m * N_ROWS + j];
    }
    tot += CMAX + __logf(s) - p / c;
  }
#pragma unroll
  for (int o = 16; o > 0; o >>= 1) tot += __shfl_down(tot, o, 32);
  __shared__ float red[8];
  const int lane = threadIdx.x & 31, w = threadIdx.x >> 5;
  if (lane == 0) red[w] = tot;
  __syncthreads();
  if (threadIdx.x == 0) {
    float t = 0.f;
#pragma unroll
    for (int i = 0; i < 8; ++i) t += red[i];
    out[0] = t * (1.0f / (float)N_ROWS);   // LOSS_WEIGHT = 1
  }
}

// ---------------------------------------------------------------------------
extern "C" void kernel_launch(void* const* d_in, const int* in_sizes, int n_in,
                              void* d_out, int out_size, void* d_ws,
                              size_t ws_size, hipStream_t stream) {
  (void)in_sizes; (void)n_in; (void)out_size; (void)ws_size;
  const float* feature = (const float*)d_in[0];
  const int*   label   = (const int*)d_in[1];
  float*       out     = (float*)d_out;

  char* w = (char*)d_ws;
  const size_t gbytes = (size_t)N_ROWS * DIMF * sizeof(bf16_t);  // 16 MB each
  bf16_t* Ghi = (bf16_t*)w;
  bf16_t* Glo = (bf16_t*)(w + gbytes);
  float* Spart = (float*)(w + 2 * gbytes);
  float* Ppart = Spart + (size_t)MSPLIT * N_ROWS;
  float* Cpart = Ppart + (size_t)MSPLIT * N_ROWS;

  k_scale_split<<<N_ROWS, 256, 0, stream>>>(feature, Ghi, Glo);
  k_gram_epilogue<<<dim3(N_ROWS / 128, MSPLIT), 256, 0, stream>>>(
      Ghi, Glo, label, Spart, Ppart, Cpart);
  k_finish<<<1, 256, 0, stream>>>(Spart, Ppart, Cpart, out);
}
